// Model_21577915695108
// MI455X (gfx1250) — matrix operand order repaired and checked by reading, not compiled, not run on hardware
//
#include <hip/hip_runtime.h>
#include <hip/hip_bf16.h>

#define MM     1000
#define NNB    32
#define JJ     32
#define NLAYERS 5
#define CC     64
#define EDIM   32
#define BB     32
#define NEDGE  (MM * NNB)      // 32000
#define NROWS  (BB * MM)       // 32000
#define HSIZE  (BB * MM * CC)  // 2,048,000
#define FC1N   512
#define FC1K   (MM * CC)       // 64000
#define FC2N   20

// padded LDS row strides (f16 units): keep rows 16B-aligned, spread banks
#define LDA_L  72    // layer GEMM tiles: 64 + 8   (row = 144B = 9*16)
#define LDK_F  264   // fc1 tiles: 256 + 8         (row = 528B = 33*16)

typedef __attribute__((ext_vector_type(16))) _Float16 v16h;
typedef __attribute__((ext_vector_type(8)))  float    v8f;

// ---------------- WMMA helpers (wave32, 16x16x32 f16 -> f32) ----------------

__device__ __forceinline__ v8f wmma_f16(v16h a, v16h b, v8f c) {
  return __builtin_amdgcn_wmma_f32_16x16x32_f16(
      /*neg_a=*/false, a, /*neg_b=*/false, b,
      /*c_mod=*/(short)0, c, /*reuse_a=*/false, /*reuse_b=*/false);
}

// A fragment: 16x32 (MxK) f16, tile is row-major [16+][ld].
// Lane l holds row M=l&15; slots = two contiguous 8-element runs:
//   K = kbase + kofs + s            (s=0..7)
//   K = kbase + 16 + kofs + (s-8)   (s=8..15),  kofs = (l>=16)*8
__device__ __forceinline__ v16h load_A_frag(const _Float16* __restrict__ tile,
                                            int ld, int lane, int kbase) {
  int m = lane & 15;
  int kofs = (lane >> 4) ? 8 : 0;
  const _Float16* p = tile + m * ld + kbase + kofs;
  v16h a;
#pragma unroll
  for (int s = 0; s < 8; ++s)  a[s] = p[s];        // 16B run
#pragma unroll
  for (int s = 0; s < 8; ++s)  a[8 + s] = p[16 + s]; // 16B run
  return a;
}

// B fragment: 32x16 (KxN) f16 from a TRANSPOSED tile wt[n][k] (N-major).
// Lane l holds column N=nbase+(l&15); same K slots as A: two 8-element runs (kofs, 16+kofs).
__device__ __forceinline__ v16h load_B_frag_t(const _Float16* __restrict__ wt,
                                              int ld, int lane, int kbase, int nbase) {
  int n = nbase + (lane & 15);
  const _Float16* p = wt + n * ld + kbase + ((lane >> 4) << 3);
  v16h b;
#pragma unroll
  for (int s = 0; s < 8; ++s) b[s] = p[s];
#pragma unroll
  for (int s = 0; s < 8; ++s) b[8 + s] = p[16 + s];
  return b;
}

// ---------------- elementwise / setup kernels ----------------

__global__ void k_transpose(const float* __restrict__ x, float* __restrict__ h) {
  int idx = blockIdx.x * blockDim.x + threadIdx.x;
  if (idx >= HSIZE) return;
  int c = idx % CC;
  int m = (idx / CC) % MM;
  int b = idx / (CC * MM);
  h[idx] = x[(b * CC + c) * MM + m];   // x: [B,C,M] -> h: [B,M,C]
}

__global__ void k_edge(const float* __restrict__ maps, const float* __restrict__ ew,
                       const float* __restrict__ eb, float* __restrict__ e) {
  int i = blockIdx.x * blockDim.x + threadIdx.x;
  if (i >= NEDGE) return;
  float acc[5];
#pragma unroll
  for (int d = 0; d < 5; ++d) acc[d] = eb[d];
  for (int k = 0; k < EDIM; ++k) {
    float mv = maps[i * EDIM + k];
#pragma unroll
    for (int d = 0; d < 5; ++d) acc[d] += mv * ew[k * 5 + d];
  }
#pragma unroll
  for (int d = 0; d < 5; ++d) e[i * 5 + d] = acc[d];
}

// One wave (32 lanes) per node: Gaussian mixture weight, softmax over 32
// neighbors via shuffles, duplicate-column kill (scatter .set semantics).
__global__ void k_attn(const float* __restrict__ e, const int* __restrict__ lidx,
                       const float* __restrict__ mu, const float* __restrict__ sigma,
                       float* __restrict__ attn) {
  __shared__ float sMu[JJ * 5];
  __shared__ float sSg[JJ * 5];
  int lane = threadIdx.x;
  int n = blockIdx.x;
  for (int t = lane; t < JJ * 5; t += 32) { sMu[t] = mu[t]; sSg[t] = sigma[t]; }
  __syncthreads();
  int edge = n * NNB + lane;
  float e5[5];
#pragma unroll
  for (int d = 0; d < 5; ++d) e5[d] = e[edge * 5 + d];
  float w = 0.f;
  for (int j = 0; j < JJ; ++j) {
    float val = 0.f;
#pragma unroll
    for (int d = 0; d < 5; ++d) {
      float u = e5[d] - sMu[j * 5 + d];
      val += u * u * sSg[j * 5 + d];
    }
    w += __expf(-0.5f * val);
  }
  float mx = w;
  for (int off = 16; off > 0; off >>= 1) mx = fmaxf(mx, __shfl_xor(mx, off, 32));
  float ex = __expf(w - mx);
  float sm = ex;
  for (int off = 16; off > 0; off >>= 1) sm += __shfl_xor(sm, off, 32);
  float a = ex / sm;
  int col = lidx[edge] % MM;
  bool dead = false;
  for (int j = 0; j < NNB; ++j) {
    int cj = __shfl(col, j, 32);
    if (j > lane && cj == col) dead = true;
  }
  attn[edge] = dead ? 0.f : a;
}

// Sparse L @ h : block = (node n, batch b), 64 threads = channels.
__global__ void k_spmm(const float* __restrict__ h, const float* __restrict__ attn,
                       const int* __restrict__ lidx, float* __restrict__ lx) {
  int n = blockIdx.x;
  int b = blockIdx.y;
  int c = threadIdx.x;
  __shared__ float sA[NNB];
  __shared__ int   sC[NNB];
  if (c < NNB) { sA[c] = attn[n * NNB + c]; sC[c] = lidx[n * NNB + c] % MM; }
  __syncthreads();
  float acc = 0.f;
#pragma unroll 8
  for (int k = 0; k < NNB; ++k)
    acc += sA[k] * h[(b * MM + sC[k]) * CC + c];
  lx[(b * MM + n) * CC + c] = acc;
}

// out = Lx@W1 + b1 + h@W2 + b2 ; 64 rows/block, 4 waves, WMMA f16->f32.
// Weights staged TRANSPOSED (N-major) into LDS for contiguous B fragments.
__global__ void k_layergemm(const float* __restrict__ lx, const float* __restrict__ h,
                            const float* __restrict__ w1, const float* __restrict__ b1,
                            const float* __restrict__ w2, const float* __restrict__ b2,
                            float* __restrict__ out) {
  __shared__ _Float16 sW1t[CC * LDA_L];
  __shared__ _Float16 sW2t[CC * LDA_L];
  __shared__ _Float16 sLx[64 * LDA_L];
  __shared__ _Float16 sH[64 * LDA_L];
  int tid = threadIdx.x;                       // 128 threads
  int row0 = blockIdx.x * 64;
  for (int i = tid; i < CC * CC; i += 128) {
    int k = i >> 6, n = i & 63;
    sW1t[n * LDA_L + k] = (_Float16)w1[i];     // transpose: [n][k]
    sW2t[n * LDA_L + k] = (_Float16)w2[i];
    sLx[k * LDA_L + n]  = (_Float16)lx[row0 * CC + i];  // row-major [m][k]
    sH[k * LDA_L + n]   = (_Float16)h[row0 * CC + i];
  }
  __syncthreads();
  int wave = tid >> 5, lane = tid & 31;
  const _Float16* myLx = sLx + wave * 16 * LDA_L;
  const _Float16* myH  = sH  + wave * 16 * LDA_L;
  v8f acc[4] = {};
#pragma unroll
  for (int kb = 0; kb < CC; kb += 32) {
    v16h a1 = load_A_frag(myLx, LDA_L, lane, kb);
    v16h a2 = load_A_frag(myH,  LDA_L, lane, kb);
#pragma unroll
    for (int t = 0; t < 4; ++t) {
      acc[t] = wmma_f16(a1, load_B_frag_t(sW1t, LDA_L, lane, kb, t * 16), acc[t]);
      acc[t] = wmma_f16(a2, load_B_frag_t(sW2t, LDA_L, lane, kb, t * 16), acc[t]);
    }
  }
  int rr = (lane >> 4) << 3;
  int nn = lane & 15;
#pragma unroll
  for (int t = 0; t < 4; ++t) {
    int gc = t * 16 + nn;
    float bias = b1[gc] + b2[gc];
#pragma unroll
    for (int r = 0; r < 8; ++r) {
      int gm = row0 + wave * 16 + rr + r;
      out[gm * CC + gc] = acc[t][r] + bias;
    }
  }
}

// ---------------- BatchNorm ----------------

__global__ void k_zero(float* p, int n) {
  int i = blockIdx.x * blockDim.x + threadIdx.x;
  if (i < n) p[i] = 0.f;
}

__global__ void k_bnstats(const float* __restrict__ h, float* __restrict__ sum,
                          float* __restrict__ sq) {
  int c = threadIdx.x;                 // 64 channels
  int r0 = blockIdx.x * 256;
  float s = 0.f, q = 0.f;
  for (int r = 0; r < 256; ++r) {
    float v = h[(r0 + r) * CC + c];
    s += v; q += v * v;
  }
  atomicAdd(&sum[c], s);
  atomicAdd(&sq[c], q);
}

__global__ void k_bnfin(const float* sum, const float* sq, const float* g,
                        const float* b, float* scale, float* shift) {
  int c = threadIdx.x;
  float mean = sum[c] * (1.f / (float)NROWS);
  float var  = sq[c] * (1.f / (float)NROWS) - mean * mean;
  float inv  = rsqrtf(var + 1e-5f);
  float sc   = g[c] * inv;
  scale[c] = sc;
  shift[c] = b[c] - mean * sc;
}

__global__ void k_bnapply(float* h, const float* __restrict__ scale,
                          const float* __restrict__ shift) {
  int i = blockIdx.x * blockDim.x + threadIdx.x;
  if (i >= HSIZE) return;
  int c = i & (CC - 1);
  h[i] = fmaxf(h[i] * scale[c] + shift[c], 0.f);
}

// ---------------- Classifier head ----------------

// fc1: [32,64000] x [64000,512] + K-split, f32 atomic accumulation.
// grid = (512/64 = 8 col blocks, 64000/256 = 250 K chunks), block = 128 (4 waves).
// Wave w: row tile (w&1), column tiles {(w>>1)*2, (w>>1)*2+1}.
__global__ void k_fc1(const float* __restrict__ h, const float* __restrict__ w,
                      float* __restrict__ part) {
  __shared__ _Float16 sA[BB * LDK_F];      // 32 rows x 256 K   (~16.9 KB)
  __shared__ _Float16 sBt[64 * LDK_F];     // 64 cols x 256 K   (~33.8 KB)
  int tid = threadIdx.x;
  int n0 = blockIdx.x * 64;
  int k0 = blockIdx.y * 256;
  // prefetch next K chunk of the dominant weight stream into GL2
  if (k0 + 256 < FC1K)
    __builtin_prefetch(&w[(size_t)(k0 + 256) * FC1N + n0 + (tid & 63)], 0, 1);
  for (int i = tid; i < BB * 256; i += 128) {
    int r = i >> 8, k = i & 255;
    sA[r * LDK_F + k] = (_Float16)h[r * FC1K + k0 + k];
  }
  for (int i = tid; i < 64 * 256; i += 128) {
    int k = i >> 6, n = i & 63;
    sBt[n * LDK_F + k] = (_Float16)w[(k0 + k) * FC1N + n0 + n];  // transpose
  }
  __syncthreads();
  int wave = tid >> 5, lane = tid & 31;
  int rowt = wave & 1;                 // 0..1  -> rows rowt*16
  int colp = wave >> 1;                // 0..1  -> cols colp*32
  const _Float16* myA = sA + rowt * 16 * LDK_F;
  v8f acc0 = {}, acc1 = {};
#pragma unroll 4
  for (int kb = 0; kb < 256; kb += 32) {
    v16h a  = load_A_frag(myA, LDK_F, lane, kb);
    acc0 = wmma_f16(a, load_B_frag_t(sBt, LDK_F, lane, kb, colp * 32), acc0);
    acc1 = wmma_f16(a, load_B_frag_t(sBt, LDK_F, lane, kb, colp * 32 + 16), acc1);
  }
  int rr = (lane >> 4) << 3;
  int n  = lane & 15;
#pragma unroll
  for (int r = 0; r < 8; ++r) {
    int row = rowt * 16 + rr + r;
    atomicAdd(&part[row * FC1N + n0 + colp * 32 + n], acc0[r]);
    atomicAdd(&part[row * FC1N + n0 + colp * 32 + 16 + n], acc1[r]);
  }
}

__global__ void k_fc1fin(const float* __restrict__ part, const float* __restrict__ bias,
                         float* __restrict__ o) {
  int i = blockIdx.x * blockDim.x + threadIdx.x;
  if (i >= BB * FC1N) return;
  o[i] = fmaxf(part[i] + bias[i & (FC1N - 1)], 0.f);
}

__global__ void k_fc2(const float* __restrict__ h, const float* __restrict__ w,
                      const float* __restrict__ bias, float* __restrict__ out) {
  int t = blockIdx.x * blockDim.x + threadIdx.x;
  if (t >= BB * FC2N) return;
  int b = t / FC2N, o = t % FC2N;
  float acc = bias[o];
  for (int k = 0; k < FC1N; ++k) acc += h[b * FC1N + k] * w[k * FC2N + o];
  out[t] = acc;
}

// ---------------- host launch ----------------

extern "C" void kernel_launch(void* const* d_in, const int* in_sizes, int n_in,
                              void* d_out, int out_size, void* d_ws, size_t ws_size,
                              hipStream_t stream) {
  (void)in_sizes; (void)n_in; (void)out_size; (void)ws_size;
  const float* x     = (const float*)d_in[0];
  const float* maps  = (const float*)d_in[1];
  const int*   lidx  = (const int*)d_in[2];
  const float* ew    = (const float*)d_in[3];
  const float* ebias = (const float*)d_in[4];
  const float* mu    = (const float*)d_in[5];
  const float* sigma = (const float*)d_in[6];
  const float* lin_w = (const float*)d_in[7];
  const float* lin_b = (const float*)d_in[8];
  const float* lin2w = (const float*)d_in[9];
  const float* lin2b = (const float*)d_in[10];
  const float* bng   = (const float*)d_in[11];
  const float* bnb   = (const float*)d_in[12];
  const float* fc1w  = (const float*)d_in[13];
  const float* fc1b  = (const float*)d_in[14];
  const float* fc2w  = (const float*)d_in[15];
  const float* fc2b  = (const float*)d_in[16];
  float* out = (float*)d_out;

  float* ws = (float*)d_ws;
  float* e      = ws;  ws += NEDGE * 5;
  float* attn   = ws;  ws += NEDGE;
  float* hA     = ws;  ws += HSIZE;
  float* hB     = ws;  ws += HSIZE;
  float* lx     = ws;  ws += HSIZE;
  float* bsum   = ws;  ws += CC;       // bsum, bsq contiguous (zeroed together)
  float* bsq    = ws;  ws += CC;
  float* bscale = ws;  ws += CC;
  float* bshift = ws;  ws += CC;
  float* fc1p   = ws;  ws += BB * FC1N;
  float* fc1o   = ws;  ws += BB * FC1N;

  k_transpose<<<(HSIZE + 255) / 256, 256, 0, stream>>>(x, hA);
  k_edge<<<(NEDGE + 255) / 256, 256, 0, stream>>>(maps, ew, ebias, e);

  float* hin = hA;
  float* hout = hB;
  for (int i = 0; i < NLAYERS; ++i) {
    k_attn<<<MM, 32, 0, stream>>>(e, lidx, mu + i * JJ * 5, sigma + i * JJ * 5, attn);
    dim3 gs(MM, BB);
    k_spmm<<<gs, CC, 0, stream>>>(hin, attn, lidx, lx);
    k_layergemm<<<NROWS / 64, 128, 0, stream>>>(
        lx, hin, lin_w + i * CC * CC, lin_b + i * CC,
        lin2w + i * CC * CC, lin2b + i * CC, hout);
    k_zero<<<1, 2 * CC, 0, stream>>>(bsum, 2 * CC);
    k_bnstats<<<NROWS / 256, CC, 0, stream>>>(hout, bsum, bsq);
    k_bnfin<<<1, CC, 0, stream>>>(bsum, bsq, bng + i * CC, bnb + i * CC, bscale, bshift);
    k_bnapply<<<(HSIZE + 255) / 256, 256, 0, stream>>>(hout, bscale, bshift);
    float* t = hin; hin = hout; hout = t;
  }

  k_zero<<<(BB * FC1N + 127) / 128, 128, 0, stream>>>(fc1p, BB * FC1N);
  dim3 gfc1(FC1N / 64, FC1K / 256);
  k_fc1<<<gfc1, 128, 0, stream>>>(hin, fc1w, fc1p);
  k_fc1fin<<<(BB * FC1N + 255) / 256, 256, 0, stream>>>(fc1p, fc1b, fc1o);
  k_fc2<<<(BB * FC2N + 255) / 256, 256, 0, stream>>>(fc1o, fc2w, fc2b, out);
}
